// cnn_lstm_attention_gt_59889023976035
// MI455X (gfx1250) — compile-verified, matrix operand order checked by reading, not run on hardware
//
#include <hip/hip_runtime.h>

// ---------------------------------------------------------------------------
// Types for CDNA5 WMMA / TDM (gfx1250, wave32)
// ---------------------------------------------------------------------------
typedef __attribute__((ext_vector_type(16))) __bf16       v16bf;
typedef __attribute__((ext_vector_type(8)))  float        v8f;
typedef __attribute__((ext_vector_type(8)))  unsigned short us8;
typedef __attribute__((ext_vector_type(4)))  unsigned int  v4u;
typedef __attribute__((ext_vector_type(8)))  int           v8i;
typedef __attribute__((ext_vector_type(4)))  int           v4i;

#define DEV static __device__ __forceinline__

DEV unsigned short f2bf(float f) {
    unsigned u = __float_as_uint(f);
    unsigned r = u + 0x7FFFu + ((u >> 16) & 1u);   // round-to-nearest-even
    return (unsigned short)(r >> 16);
}
DEV float bf2f(unsigned short h) { return __uint_as_float(((unsigned)h) << 16); }
DEV float sigf(float x) { return 1.f / (1.f + expf(-x)); }

union FragBF { us8 h[2]; v16bf v; };
static_assert(sizeof(FragBF) == 32, "frag size");

// A-fragment (16x32 bf16, row-major src, row = lane&15):
//   lane<16 : elems 0..7 = K 0..7,  elems 8..15 = K 16..23
//   lane>=16: elems 0..7 = K 8..15, elems 8..15 = K 24..31
// B uses the identical per-lane layout with "row" = output column n,
// reading W[n][k] (i.e. C = A @ W^T), so both loads are contiguous b128s.
DEV v16bf load_frag(const unsigned short* base, int ld, int k0, int lane) {
    int r  = lane & 15;
    int ks = (lane >> 4) << 3;                 // 0 or 8
    const unsigned short* p = base + (size_t)r * ld + k0 + ks;
    FragBF f;
    f.h[0] = *(const us8*)(p);                 // K = ks .. ks+7
    f.h[1] = *(const us8*)(p + 16);            // K = ks+16 .. ks+23
    return f.v;
}

// ---------------------------------------------------------------------------
// Generic WMMA GEMM: C[M x N] = A(M x K, bf16) @ W(N x K, bf16)^T  (+epilogue)
// grid = (N/16, M/16, batch); block = 32 (one wave per 16x16 tile)
// mode: 0 none, 1 tanh, 2 gelu(exact), 3 sim-epilogue, 4 attnV-epilogue
// ---------------------------------------------------------------------------
__global__ void k_gemm(const unsigned short* __restrict__ A, int lda, long long bA,
                       const unsigned short* __restrict__ B, int ldb, long long bB,
                       float* __restrict__ C, int ldc, long long bC,
                       unsigned short* __restrict__ Cbf, int ldcbf, long long bCbf,
                       const float* __restrict__ bias,
                       const float* __restrict__ addmat, int ldadd,
                       int K, int mode,
                       const float* __restrict__ aux0,
                       const float* __restrict__ aux1,
                       const float* __restrict__ aux2,
                       float scale)
{
    int lane = threadIdx.x;
    int n0 = blockIdx.x * 16, m0 = blockIdx.y * 16;
    int z  = blockIdx.z;
    const unsigned short* Az = A + (long long)z * bA + (size_t)m0 * lda;
    const unsigned short* Bz = B + (long long)z * bB + (size_t)n0 * ldb;

    __builtin_prefetch(Az, 0, 1);              // global_prefetch_b8
    __builtin_prefetch(Bz, 0, 1);

    v8f acc = {0.f, 0.f, 0.f, 0.f, 0.f, 0.f, 0.f, 0.f};
    for (int k0 = 0; k0 < K; k0 += 32) {
        v16bf fa = load_frag(Az, lda, k0, lane);
        v16bf fb = load_frag(Bz, ldb, k0, lane);
        acc = __builtin_amdgcn_wmma_f32_16x16x32_bf16(false, fa, false, fb,
                                                      (short)0, acc, false, false);
    }

    int cn = lane & 15;
    int mb = (lane >> 4) * 8;
    int n  = n0 + cn;
    float*          Cz  = C   ? C   + (long long)z * bC   : nullptr;
    unsigned short* Cbz = Cbf ? Cbf + (long long)z * bCbf : nullptr;

#pragma unroll
    for (int r = 0; r < 8; r++) {
        int m = m0 + mb + r;
        float v = acc[r];
        if (bias)   v += bias[n];
        if (addmat) v += addmat[(size_t)m * ldadd + n];
        if (mode == 1)      v = tanhf(v);
        else if (mode == 2) v = 0.5f * v * (1.f + erff(v * 0.70710678118654752f));
        else if (mode == 3) // sim: (qk + adj*qew[h,i] + qeb[h,i]) * SCALE
            v = (v + aux0[(size_t)m * ldc + n] * aux1[z * 256 + m]
                   + aux2[z * 256 + m]) * scale;
        else if (mode == 4) // attnV: + e_w[h,d]*aw[h,i] + e_b[h,d]
            v = v + aux0[z * 64 + n] * aux2[z * 256 + m] + aux1[z * 64 + n];
        if (Cz)  Cz[(size_t)m * ldc + n] = v;
        if (Cbz) Cbz[(size_t)m * ldcbf + n] = f2bf(v);
    }
}

// ---------------------------------------------------------------------------
// Pack conv weights (80,80,3) -> 3 shifted GEMM operands Wk (3, 80, 96) bf16
// ---------------------------------------------------------------------------
__global__ void k_cvt_convw(const float* __restrict__ src, unsigned short* __restrict__ dst) {
    int i = blockIdx.x * 256 + threadIdx.x;
    if (i >= 3 * 80 * 96) return;
    int k = i / (80 * 96);
    int r = i % (80 * 96);
    int oc = r / 96, ic = r % 96;
    dst[i] = f2bf(ic < 80 ? src[(oc * 80 + ic) * 3 + k] : 0.f);
}

// ---------------------------------------------------------------------------
// Fused conv1 -> conv2(WMMA) -> conv3(WMMA) -> mean, one block per sample.
// Input tile staged into LDS by the Tensor Data Mover (TENSOR_LOAD_TO_LDS,
// TENSORcnt-tracked) while all waves zero the activation buffers.
// A k-tap conv is a sum of 3 shifted GEMMs: out[:,t] = sum_k Wk @ x[:,t-2+k].
// Activations kept transposed in LDS ([t+2][ch], bf16, zero halo) so the
// B-fragment for output column t is a contiguous LDS row at (n0+k)*96.
// xss: (1536, 7, 128). Output xfeat (1536,80) fp32 + bf16 padded to 96 cols.
// ---------------------------------------------------------------------------
__global__ void k_conv(const float* __restrict__ xss,
                       const float* __restrict__ w1, const float* __restrict__ b1,
                       const unsigned short* __restrict__ w2k, const float* __restrict__ b2,
                       const unsigned short* __restrict__ w3k, const float* __restrict__ b3,
                       float* __restrict__ xfeat, unsigned short* __restrict__ xfeat_bf)
{
    __shared__ __align__(16) float          s_in[7 * 128];
    __shared__ __align__(16) unsigned short s_x1[146 * 96];  // conv1 out, row = t1+2
    __shared__ __align__(16) unsigned short s_x2[146 * 96];  // conv2 out, row = t2+2
    __shared__ float s_sum[80];

    int s = blockIdx.x, tid = threadIdx.x;
    int wave = tid >> 5, lane = tid & 31;

    if (wave == 0) {
        // TDM descriptor (D#): 2D tile 128x7 of 4-byte data, global -> LDS
        unsigned lds_addr = (unsigned)(size_t)(void*)s_in;
        unsigned long long ga = (unsigned long long)(size_t)(xss + (size_t)s * 7 * 128);
        v4u g0;
        g0.x = 1u;                                            // count=1 (valid, user mode)
        g0.y = lds_addr;                                      // lds_addr[31:0]
        g0.z = (unsigned)(ga & 0xFFFFFFFFu);                  // global_addr[31:0]
        g0.w = (unsigned)((ga >> 32) & 0x01FFFFFFu) | (2u << 30); // addr[56:32] | type=2
        v8i g1;
        g1[0] = (int)(2u << 16);                 // wg_mask=0, data_size=2 (4 bytes)
        g1[1] = (int)(128u << 16);               // tensor_dim0[15:0] = 128
        g1[2] = (int)(7u << 16);                 // tdim0 hi=0 | tensor_dim1[15:0] = 7
        g1[3] = (int)(128u << 16);               // tdim1 hi=0 | tile_dim0 = 128
        g1[4] = (int)7;                          // tile_dim1 = 7, tile_dim2 = 0
        g1[5] = (int)128;                        // tensor_dim0_stride[31:0] = 128 elems
        g1[6] = 0;                               // stride0 hi | stride1 lo (unused)
        g1[7] = 0;
        v4i gz4 = {0, 0, 0, 0};
        v8i gz8 = {0, 0, 0, 0, 0, 0, 0, 0};
        __builtin_amdgcn_tensor_load_to_lds(g0, g1, gz4, gz4, gz8, 0);
    }
    for (int i = tid; i < 146 * 96; i += 256) { s_x1[i] = 0; s_x2[i] = 0; }
    if (tid < 80) s_sum[tid] = 0.f;
    if (wave == 0) __builtin_amdgcn_s_wait_tensorcnt(0);      // s_wait_tensorcnt 0
    __syncthreads();

    // conv1 (VALU, K=21 only): (7,128) -> (80,130), relu, transposed store
    for (int idx = tid; idx < 80 * 130; idx += 256) {
        int oc = idx / 130, t = idx % 130;
        float acc = b1[oc];
        for (int ic = 0; ic < 7; ic++) {
            const float* wp = w1 + (oc * 7 + ic) * 3;
            int ti = t - 2;
#pragma unroll
            for (int k = 0; k < 3; k++, ti++)
                if (ti >= 0 && ti < 128) acc += wp[k] * s_in[ic * 128 + ti];
        }
        s_x1[(t + 2) * 96 + oc] = f2bf(fmaxf(acc, 0.f));
    }
    __syncthreads();

    // conv2 via WMMA: out (80 x 132) = sum_k W2k(80x96) @ X1^T-shift, relu
    // 5 M-tiles x 9 N-tiles = 45 tiles, round-robin over 8 waves (wave-uniform)
    for (int tile = wave; tile < 45; tile += 8) {
        int m0 = (tile / 9) * 16, n0 = (tile % 9) * 16;
        int mb = (lane >> 4) * 8, cn = lane & 15;
        v8f acc;
#pragma unroll
        for (int r = 0; r < 8; r++) acc[r] = b2[m0 + mb + r];
#pragma unroll
        for (int k = 0; k < 3; k++) {
            const unsigned short* Ab = w2k + (size_t)k * 80 * 96 + m0 * 96;
            const unsigned short* Bb = s_x1 + (n0 + k) * 96;
#pragma unroll
            for (int kc = 0; kc < 96; kc += 32) {
                v16bf fa = load_frag(Ab, 96, kc, lane);
                v16bf fb = load_frag(Bb, 96, kc, lane);
                acc = __builtin_amdgcn_wmma_f32_16x16x32_bf16(false, fa, false, fb,
                                                              (short)0, acc, false, false);
            }
        }
        int t2 = n0 + cn;
        if (t2 < 132) {
#pragma unroll
            for (int r = 0; r < 8; r++)
                s_x2[(t2 + 2) * 96 + m0 + mb + r] = f2bf(fmaxf(acc[r], 0.f));
        }
    }
    __syncthreads();

    // conv3 via WMMA: out (80 x 134), relu, then mean over t into s_sum
    for (int tile = wave; tile < 45; tile += 8) {
        int m0 = (tile / 9) * 16, n0 = (tile % 9) * 16;
        int mb = (lane >> 4) * 8, cn = lane & 15;
        v8f acc;
#pragma unroll
        for (int r = 0; r < 8; r++) acc[r] = b3[m0 + mb + r];
#pragma unroll
        for (int k = 0; k < 3; k++) {
            const unsigned short* Ab = w3k + (size_t)k * 80 * 96 + m0 * 96;
            const unsigned short* Bb = s_x2 + (n0 + k) * 96;
#pragma unroll
            for (int kc = 0; kc < 96; kc += 32) {
                v16bf fa = load_frag(Ab, 96, kc, lane);
                v16bf fb = load_frag(Bb, 96, kc, lane);
                acc = __builtin_amdgcn_wmma_f32_16x16x32_bf16(false, fa, false, fb,
                                                              (short)0, acc, false, false);
            }
        }
        int t3 = n0 + cn;
        if (t3 < 134) {
#pragma unroll
            for (int r = 0; r < 8; r++)
                atomicAdd(&s_sum[m0 + mb + r], fmaxf(acc[r], 0.f) * (1.f / 134.f));
        }
    }
    __syncthreads();

    if (tid < 96) {
        float v = (tid < 80) ? s_sum[tid] : 0.f;
        if (tid < 80) xfeat[(size_t)s * 80 + tid] = v;
        xfeat_bf[(size_t)s * 96 + tid] = f2bf(v);
    }
}

// ---------------------------------------------------------------------------
// Helpers: f32->bf16 copy / padded copy / transposed copy / vector add
// ---------------------------------------------------------------------------
__global__ void k_cvt(const float* __restrict__ s, unsigned short* __restrict__ d, int n) {
    int i = blockIdx.x * 256 + threadIdx.x;
    if (i < n) d[i] = f2bf(s[i]);
}
__global__ void k_cvt_pad(const float* __restrict__ s, unsigned short* __restrict__ d,
                          int rows, int K, int Kp) {
    int i = blockIdx.x * 256 + threadIdx.x;
    if (i >= rows * Kp) return;
    int r = i / Kp, k = i % Kp;
    d[i] = f2bf(k < K ? s[r * K + k] : 0.f);
}
__global__ void k_cvtT(const float* __restrict__ s, unsigned short* __restrict__ d,
                       int N, int K) { // d[n*K+k] = s[k*N+n]
    int i = blockIdx.x * 256 + threadIdx.x;
    if (i >= N * K) return;
    int n = i / K, k = i % K;
    d[i] = f2bf(s[k * N + n]);
}
__global__ void k_addvec(const float* a, const float* b, float* o, int n) {
    int i = blockIdx.x * 256 + threadIdx.x;
    if (i < n) o[i] = a[i] + b[i];
}

// ---------------------------------------------------------------------------
// h0/c0 init: h = static@hid_w^T + hid_b, c likewise (K=129, tiny -> VALU)
// ---------------------------------------------------------------------------
__global__ void k_hc_init(const float* __restrict__ stat,
                          const float* __restrict__ hw, const float* __restrict__ hb,
                          const float* __restrict__ cw, const float* __restrict__ cb,
                          float* h_f, float* h_r, float* c_f, float* c_r,
                          unsigned short* hbf_f, unsigned short* hbf_r)
{
    int n = blockIdx.x, j = threadIdx.x; // 160 threads
    const float* sr = stat + n * 129;
    float hv = hb[j], cv = cb[j];
    for (int k = 0; k < 129; k++) {
        float sv = sr[k];
        hv += sv * hw[j * 129 + k];
        cv += sv * cw[j * 129 + k];
    }
    if (j < 80) { h_f[n * 80 + j] = hv; c_f[n * 80 + j] = cv; hbf_f[n * 96 + j] = f2bf(hv); }
    else        { h_r[n * 80 + j - 80] = hv; c_r[n * 80 + j - 80] = cv; hbf_r[n * 96 + j - 80] = f2bf(hv); }
    if (j < 16) { hbf_f[n * 96 + 80 + j] = 0; hbf_r[n * 96 + 80 + j] = 0; }
}

// LSTM cell pointwise update
__global__ void k_lstm_cell(const float* __restrict__ gates, float* __restrict__ c,
                            float* __restrict__ lout, int col_off, int t,
                            unsigned short* __restrict__ hbf)
{
    int idx = blockIdx.x * blockDim.x + threadIdx.x;
    if (idx >= 256 * 80) return;
    int n = idx / 80, j = idx % 80;
    const float* g = gates + n * 320;
    float ig = sigf(g[j]), fg = sigf(g[80 + j]);
    float gg = tanhf(g[160 + j]), og = sigf(g[240 + j]);
    float cv = fg * c[idx] + ig * gg;
    c[idx] = cv;
    float hv = og * tanhf(cv);
    lout[(size_t)(n * 6 + t) * 160 + col_off + j] = hv;
    hbf[n * 96 + j] = f2bf(hv);
}

// attention pooling over T=6
__global__ void k_pool(const float* __restrict__ u, const float* __restrict__ uw,
                       const float* __restrict__ lout,
                       float* __restrict__ feat, unsigned short* __restrict__ feat_bf)
{
    __shared__ float s_red[160];
    __shared__ float s_sc[6];
    int n = blockIdx.x, d = threadIdx.x;
    for (int t = 0; t < 6; t++) {
        s_red[d] = u[(size_t)(n * 6 + t) * 160 + d] * uw[d];
        __syncthreads();
        if (d == 0) { float s = 0; for (int i = 0; i < 160; i++) s += s_red[i]; s_sc[t] = s; }
        __syncthreads();
    }
    if (d == 0) {
        float m = s_sc[0];
        for (int t = 1; t < 6; t++) m = fmaxf(m, s_sc[t]);
        float s = 0;
        for (int t = 0; t < 6; t++) { s_sc[t] = expf(s_sc[t] - m); s += s_sc[t]; }
        for (int t = 0; t < 6; t++) s_sc[t] /= s;
    }
    __syncthreads();
    float f = 0;
    for (int t = 0; t < 6; t++) f += lout[(size_t)(n * 6 + t) * 160 + d] * s_sc[t];
    feat[n * 160 + d] = f;
    feat_bf[n * 160 + d] = f2bf(f);
}

// graph build
__global__ void k_pn(const float* __restrict__ gram, float* __restrict__ pn) {
    int i = threadIdx.x;
    pn[i] = sqrtf(gram[i * 257]);
}
__global__ void k_dinv(const float* __restrict__ gram, const float* __restrict__ pn,
                       float* __restrict__ dinv) {
    __shared__ float s[256];
    int i = blockIdx.x, j = threadIdx.x;
    float pc = gram[(size_t)i * 256 + j] / (pn[i] * pn[j]);
    s[j] = (pc > 0.70710678118654752f) ? pc : 0.f;
    __syncthreads();
    if (j == 0) { float t = 0; for (int k = 0; k < 256; k++) t += s[k]; dinv[i] = rsqrtf(t); }
}
__global__ void k_adj(const float* __restrict__ gram, const float* __restrict__ pn,
                      const float* __restrict__ dinv, float* __restrict__ adj) {
    int i = blockIdx.x, j = threadIdx.x;
    float pc = gram[(size_t)i * 256 + j] / (pn[i] * pn[j]);
    float a = (pc > 0.70710678118654752f) ? pc : 0.f;
    adj[(size_t)i * 256 + j] = dinv[i] * a * dinv[j];
}

// layernorm(160) -> bf16
__global__ void k_ln(const float* __restrict__ x, const float* __restrict__ g,
                     const float* __restrict__ b, unsigned short* __restrict__ out) {
    __shared__ float s[160];
    __shared__ float mv[2];
    int n = blockIdx.x, d = threadIdx.x;
    float v = x[n * 160 + d];
    s[d] = v; __syncthreads();
    if (d == 0) { float m = 0; for (int i = 0; i < 160; i++) m += s[i]; mv[0] = m / 160.f; }
    __syncthreads();
    float c = v - mv[0];
    s[d] = c * c; __syncthreads();
    if (d == 0) { float m = 0; for (int i = 0; i < 160; i++) m += s[i]; mv[1] = rsqrtf(m / 160.f + 1e-5f); }
    __syncthreads();
    out[n * 160 + d] = f2bf(c * mv[1] * g[d] + b[d]);
}

// rotary on q,k  + per-(h,i) edge dot products  (rank-1 edge trick)
__global__ void k_rot(const float* __restrict__ q, const float* __restrict__ kv,
                      const float* __restrict__ ew, const float* __restrict__ eb,
                      unsigned short* __restrict__ qbf, unsigned short* __restrict__ kbf,
                      float* __restrict__ qew, float* __restrict__ qeb)
{
    __shared__ float s[64];
    int i = blockIdx.x, h = blockIdx.y, d = threadIdx.x;
    float fr = (float)i * powf(10000.f, -(float)(d & ~1) / 64.f);
    float cs = cosf(fr), sn = sinf(fr);
    int qb = i * 512 + h * 64;
    float qa = q[qb + d], qo = q[qb + (d ^ 1)];
    float qr = (d & 1) ? (qa * cs + qo * sn) : (qa * cs - qo * sn);
    int kb = i * 1024 + h * 64;
    float ka = kv[kb + d], ko = kv[kb + (d ^ 1)];
    float kr = (d & 1) ? (ka * cs + ko * sn) : (ka * cs - ko * sn);
    qbf[qb + d] = f2bf(qr);
    kbf[qb + d] = f2bf(kr);
    s[d] = qr * ew[h * 64 + d]; __syncthreads();
    if (d == 0) { float t = 0; for (int k = 0; k < 64; k++) t += s[k]; qew[h * 256 + i] = t; }
    __syncthreads();
    s[d] = qr * eb[h * 64 + d]; __syncthreads();
    if (d == 0) { float t = 0; for (int k = 0; k < 64; k++) t += s[k]; qeb[h * 256 + i] = t; }
}

// v^T (per head) to bf16 for attn@v GEMM operand
__global__ void k_vT(const float* __restrict__ kv, unsigned short* __restrict__ vT) {
    int idx = blockIdx.x * 256 + threadIdx.x;   // idx = hd*256 + j, 512 blocks
    if (idx >= 512 * 256) return;
    int hd = idx >> 8, j = idx & 255;
    vT[idx] = f2bf(kv[(size_t)j * 1024 + 512 + hd]);
}

// row softmax over 256, emit bf16 attn + aw = sum(attn*adj)
__global__ void k_softmax(const float* __restrict__ sim, const float* __restrict__ adj,
                          unsigned short* __restrict__ attn, float* __restrict__ aw)
{
    __shared__ float s[256];
    __shared__ float red[2];
    int b = blockIdx.x, j = threadIdx.x;  // b = h*256 + i
    int i = b & 255;
    float v = sim[(size_t)b * 256 + j];
    s[j] = v; __syncthreads();
    if (j == 0) { float m = -1e30f; for (int k = 0; k < 256; k++) m = fmaxf(m, s[k]); red[0] = m; }
    __syncthreads();
    float e = expf(v - red[0]);
    s[j] = e; __syncthreads();
    if (j == 0) { float t = 0; for (int k = 0; k < 256; k++) t += s[k]; red[1] = t; }
    __syncthreads();
    float p = e / red[1];
    attn[(size_t)b * 256 + j] = f2bf(p);
    s[j] = p * adj[(size_t)i * 256 + j]; __syncthreads();
    if (j == 0) { float t = 0; for (int k = 0; k < 256; k++) t += s[k]; aw[b] = t; }
}

// gated residual (scalar gate per node), in-place on nodes
__global__ void k_gate(const float* __restrict__ x, float* __restrict__ nodes,
                       const float* __restrict__ w) {
    __shared__ float s[160];
    __shared__ float gg;
    int n = blockIdx.x, d = threadIdx.x;
    float xv = x[n * 160 + d], rv = nodes[n * 160 + d];
    s[d] = xv * w[d] + rv * w[160 + d] + (xv - rv) * w[320 + d];
    __syncthreads();
    if (d == 0) { float t = 0; for (int i = 0; i < 160; i++) t += s[i]; gg = sigf(t); }
    __syncthreads();
    nodes[n * 160 + d] = xv * gg + rv * (1.f - gg);
}

// final head: fc1 (160->2), elu, log_softmax
__global__ void k_head(const float* __restrict__ nodes, const float* __restrict__ w,
                       const float* __restrict__ b, float* __restrict__ out) {
    int n = threadIdx.x;
    float l0 = b[0], l1 = b[1];
    for (int d = 0; d < 160; d++) {
        float v = nodes[n * 160 + d];
        l0 += v * w[d];
        l1 += v * w[160 + d];
    }
    l0 = l0 > 0.f ? l0 : expm1f(l0);
    l1 = l1 > 0.f ? l1 : expm1f(l1);
    float m = fmaxf(l0, l1);
    float lse = m + logf(expf(l0 - m) + expf(l1 - m));
    out[n * 2 + 0] = l0 - lse;
    out[n * 2 + 1] = l1 - lse;
}

// ---------------------------------------------------------------------------
// Host orchestration
// ---------------------------------------------------------------------------
extern "C" void kernel_launch(void* const* d_in, const int* in_sizes, int n_in,
                              void* d_out, int out_size, void* d_ws, size_t ws_size,
                              hipStream_t stream)
{
    (void)in_sizes; (void)n_in; (void)out_size; (void)ws_size;
    auto P = [&](int i) { return (const float*)d_in[i]; };
    const float* xss  = P(0);
    const float* stat = P(1);

    // param indices (setup_inputs insertion order, params flattened recursively)
    enum { I_C1W = 2, I_C1B, I_C2W, I_C2B, I_C3W, I_C3B,
           I_HIDW, I_HIDB, I_CELW, I_CELB,
           I_WIHF, I_BIHF, I_WHHF, I_BHHF,
           I_WIHR, I_BIHR, I_WHHR, I_BHHR,
           I_WOM, I_UOM, I_L0 };
    enum { LP_LN1G = 0, LP_LN1B, LP_QW, LP_QB, LP_KVW, LP_KVB, LP_EW, LP_EB,
           LP_OW, LP_OB, LP_G1W, LP_LN2G, LP_LN2B, LP_F1W, LP_F1B, LP_F2W, LP_F2B, LP_G2W,
           LSTRIDE };
    const int I_FC1W = I_L0 + 2 * LSTRIDE;
    const int I_FC1B = I_FC1W + 1;

    // workspace bump allocator
    char* W = (char*)d_ws;
    size_t off = 0;
    auto alloc = [&](size_t bytes) -> void* {
        off = (off + 255) & ~(size_t)255;
        void* p = W + off;
        off += bytes;
        return p;
    };
    float*          xfeat    = (float*)alloc((size_t)1536 * 80 * 4);
    unsigned short* xfeat_bf = (unsigned short*)alloc((size_t)1536 * 96 * 2);
    unsigned short* w2k_bf   = (unsigned short*)alloc(3 * 80 * 96 * 2);
    unsigned short* w3k_bf   = (unsigned short*)alloc(3 * 80 * 96 * 2);
    unsigned short* wihf_bf  = (unsigned short*)alloc(320 * 96 * 2);
    unsigned short* whhf_bf  = (unsigned short*)alloc(320 * 96 * 2);
    unsigned short* wihr_bf  = (unsigned short*)alloc(320 * 96 * 2);
    unsigned short* whhr_bf  = (unsigned short*)alloc(320 * 96 * 2);
    float*          bsum_f   = (float*)alloc(320 * 4);
    float*          bsum_r   = (float*)alloc(320 * 4);
    float*          xw_f     = (float*)alloc((size_t)1536 * 320 * 4);
    float*          xw_r     = (float*)alloc((size_t)1536 * 320 * 4);
    float*          h_f      = (float*)alloc(256 * 80 * 4);
    float*          h_r      = (float*)alloc(256 * 80 * 4);
    float*          c_f      = (float*)alloc(256 * 80 * 4);
    float*          c_r      = (float*)alloc(256 * 80 * 4);
    unsigned short* hbf_f    = (unsigned short*)alloc(256 * 96 * 2);
    unsigned short* hbf_r    = (unsigned short*)alloc(256 * 96 * 2);
    float*          gates    = (float*)alloc(256 * 320 * 4);
    float*          lout     = (float*)alloc((size_t)1536 * 160 * 4);
    unsigned short* lout_bf  = (unsigned short*)alloc((size_t)1536 * 160 * 2);
    float*          ubuf     = (float*)alloc((size_t)1536 * 160 * 4);
    unsigned short* woT      = (unsigned short*)alloc(160 * 160 * 2);
    float*          nodes    = (float*)alloc(256 * 160 * 4);   // feat == nodes
    unsigned short* feat_bf  = (unsigned short*)alloc(256 * 160 * 2);
    float*          gram     = (float*)alloc(256 * 256 * 4);
    float*          pn       = (float*)alloc(256 * 4);
    float*          dinv     = (float*)alloc(256 * 4);
    unsigned short* lw_bf[2][5];
    const size_t lw_sz[5] = { 512 * 160, 1024 * 160, 160 * 512, 640 * 160, 160 * 640 };
    for (int l = 0; l < 2; l++)
        for (int w = 0; w < 5; w++)
            lw_bf[l][w] = (unsigned short*)alloc(lw_sz[w] * 2);
    unsigned short* xln_bf   = (unsigned short*)alloc(256 * 160 * 2);
    float*          qbuf     = (float*)alloc((size_t)256 * 512 * 4);
    float*          kvbuf    = (float*)alloc((size_t)256 * 1024 * 4);
    unsigned short* qbf      = (unsigned short*)alloc((size_t)256 * 512 * 2);
    unsigned short* kbf      = (unsigned short*)alloc((size_t)256 * 512 * 2);
    unsigned short* vT       = (unsigned short*)alloc((size_t)512 * 256 * 2);
    float*          qew      = (float*)alloc(8 * 256 * 4);
    float*          qeb      = (float*)alloc(8 * 256 * 4);
    float*          sim      = (float*)alloc((size_t)8 * 256 * 256 * 4);
    unsigned short* attn     = (unsigned short*)alloc((size_t)8 * 256 * 256 * 2);
    float*          aw       = (float*)alloc(8 * 256 * 4);
    unsigned short* ao_bf    = (unsigned short*)alloc((size_t)256 * 512 * 2);
    float*          xres     = (float*)alloc(256 * 160 * 4);
    unsigned short* ff1_bf   = (unsigned short*)alloc((size_t)256 * 640 * 2);

    float* out_logits = (float*)d_out;
    float* adj        = (float*)d_out + 512;   // edges output, also used in-layer

    auto gemm = [&](const unsigned short* A, int lda, long long bA,
                    const unsigned short* B, int ldb, long long bB,
                    float* C, int ldc, long long bC,
                    unsigned short* Cbf, int ldcbf, long long bCbf,
                    const float* bias, const float* addmat, int ldadd,
                    int M, int N, int K, int batch, int mode,
                    const float* a0, const float* a1, const float* a2, float scale) {
        dim3 g((unsigned)(N / 16), (unsigned)(M / 16), (unsigned)batch);
        k_gemm<<<g, 32, 0, stream>>>(A, lda, bA, B, ldb, bB, C, ldc, bC,
                                     Cbf, ldcbf, bCbf, bias, addmat, ldadd,
                                     K, mode, a0, a1, a2, scale);
    };
    auto cvt = [&](const float* s, unsigned short* d, int n) {
        k_cvt<<<(n + 255) / 256, 256, 0, stream>>>(s, d, n);
    };

    // ---- weight conversions ----
    k_cvt_convw<<<(3 * 80 * 96 + 255) / 256, 256, 0, stream>>>(P(I_C2W), w2k_bf);
    k_cvt_convw<<<(3 * 80 * 96 + 255) / 256, 256, 0, stream>>>(P(I_C3W), w3k_bf);
    k_cvt_pad<<<(320 * 96 + 255) / 256, 256, 0, stream>>>(P(I_WIHF), wihf_bf, 320, 80, 96);
    k_cvt_pad<<<(320 * 96 + 255) / 256, 256, 0, stream>>>(P(I_WHHF), whhf_bf, 320, 80, 96);
    k_cvt_pad<<<(320 * 96 + 255) / 256, 256, 0, stream>>>(P(I_WIHR), wihr_bf, 320, 80, 96);
    k_cvt_pad<<<(320 * 96 + 255) / 256, 256, 0, stream>>>(P(I_WHHR), whhr_bf, 320, 80, 96);
    k_cvtT<<<(160 * 160 + 255) / 256, 256, 0, stream>>>(P(I_WOM), woT, 160, 160);
    for (int l = 0; l < 2; l++) {
        int base = I_L0 + l * LSTRIDE;
        cvt(P(base + LP_QW),  lw_bf[l][0], 512 * 160);
        cvt(P(base + LP_KVW), lw_bf[l][1], 1024 * 160);
        cvt(P(base + LP_OW),  lw_bf[l][2], 160 * 512);
        cvt(P(base + LP_F1W), lw_bf[l][3], 640 * 160);
        cvt(P(base + LP_F2W), lw_bf[l][4], 160 * 640);
    }
    k_addvec<<<2, 256, 0, stream>>>(P(I_BIHF), P(I_BHHF), bsum_f, 320);
    k_addvec<<<2, 256, 0, stream>>>(P(I_BIHR), P(I_BHHR), bsum_r, 320);

    // ---- conv stack (TDM input staging, WMMA for conv2/conv3) ----
    k_conv<<<1536, 256, 0, stream>>>(xss, P(I_C1W), P(I_C1B), w2k_bf, P(I_C2B),
                                     w3k_bf, P(I_C3B), xfeat, xfeat_bf);

    // ---- x @ w_ih^T precompute (both directions), biases folded ----
    gemm(xfeat_bf, 96, 0, wihf_bf, 96, 0, xw_f, 320, 0, nullptr, 0, 0,
         bsum_f, nullptr, 0, 1536, 320, 96, 1, 0, nullptr, nullptr, nullptr, 0.f);
    gemm(xfeat_bf, 96, 0, wihr_bf, 96, 0, xw_r, 320, 0, nullptr, 0, 0,
         bsum_r, nullptr, 0, 1536, 320, 96, 1, 0, nullptr, nullptr, nullptr, 0.f);

    // ---- h0/c0 ----
    k_hc_init<<<256, 160, 0, stream>>>(stat, P(I_HIDW), P(I_HIDB), P(I_CELW), P(I_CELB),
                                       h_f, h_r, c_f, c_r, hbf_f, hbf_r);

    // ---- LSTM recurrence (6 steps, both directions) ----
    for (int s = 0; s < 6; s++) {
        gemm(hbf_f, 96, 0, whhf_bf, 96, 0, gates, 320, 0, nullptr, 0, 0,
             nullptr, xw_f + s * 320, 1920, 256, 320, 96, 1, 0,
             nullptr, nullptr, nullptr, 0.f);
        k_lstm_cell<<<80, 256, 0, stream>>>(gates, c_f, lout, 0, s, hbf_f);

        int tr = 5 - s;
        gemm(hbf_r, 96, 0, whhr_bf, 96, 0, gates, 320, 0, nullptr, 0, 0,
             nullptr, xw_r + tr * 320, 1920, 256, 320, 96, 1, 0,
             nullptr, nullptr, nullptr, 0.f);
        k_lstm_cell<<<80, 256, 0, stream>>>(gates, c_r, lout, 80, tr, hbf_r);
    }

    // ---- attention pooling ----
    cvt(lout, lout_bf, 1536 * 160);
    gemm(lout_bf, 160, 0, woT, 160, 0, ubuf, 160, 0, nullptr, 0, 0,
         nullptr, nullptr, 0, 1536, 160, 160, 1, 1, nullptr, nullptr, nullptr, 0.f);
    k_pool<<<256, 160, 0, stream>>>(ubuf, P(I_UOM), lout, nodes, feat_bf);

    // ---- graph build (gram via WMMA, pn from diagonal) ----
    gemm(feat_bf, 160, 0, feat_bf, 160, 0, gram, 256, 0, nullptr, 0, 0,
         nullptr, nullptr, 0, 256, 256, 160, 1, 0, nullptr, nullptr, nullptr, 0.f);
    k_pn<<<1, 256, 0, stream>>>(gram, pn);
    k_dinv<<<256, 256, 0, stream>>>(gram, pn, dinv);
    k_adj<<<256, 256, 0, stream>>>(gram, pn, dinv, adj);   // edges output written here

    // ---- graph transformer layers ----
    for (int l = 0; l < 2; l++) {
        int base = I_L0 + l * LSTRIDE;
        k_ln<<<256, 160, 0, stream>>>(nodes, P(base + LP_LN1G), P(base + LP_LN1B), xln_bf);
        gemm(xln_bf, 160, 0, lw_bf[l][0], 160, 0, qbuf, 512, 0, nullptr, 0, 0,
             P(base + LP_QB), nullptr, 0, 256, 512, 160, 1, 0,
             nullptr, nullptr, nullptr, 0.f);
        gemm(xln_bf, 160, 0, lw_bf[l][1], 160, 0, kvbuf, 1024, 0, nullptr, 0, 0,
             P(base + LP_KVB), nullptr, 0, 256, 1024, 160, 1, 0,
             nullptr, nullptr, nullptr, 0.f);
        k_rot<<<dim3(256, 8), 64, 0, stream>>>(qbuf, kvbuf, P(base + LP_EW), P(base + LP_EB),
                                               qbf, kbf, qew, qeb);
        k_vT<<<512, 256, 0, stream>>>(kvbuf, vT);
        // sim = (q k^T + adj*qew + qeb) * SCALE, batched over 8 heads
        gemm(qbf, 512, 64, kbf, 512, 64, sim, 256, 65536, nullptr, 0, 0,
             nullptr, nullptr, 0, 256, 256, 64, 8, 3, adj, qew, qeb, 0.125f);
        k_softmax<<<2048, 256, 0, stream>>>(sim, adj, attn, aw);
        // out = attn @ v + e_w*aw + e_b  -> bf16 (i, h*64+d)
        gemm(attn, 256, 65536, vT, 256, 64 * 256, nullptr, 0, 0,
             ao_bf, 512, 64, nullptr, nullptr, 0, 256, 64, 256, 8, 4,
             P(base + LP_EW), P(base + LP_EB), aw, 0.f);
        gemm(ao_bf, 512, 0, lw_bf[l][2], 512, 0, xres, 160, 0, nullptr, 0, 0,
             P(base + LP_OB), nullptr, 0, 256, 160, 512, 1, 0,
             nullptr, nullptr, nullptr, 0.f);
        k_gate<<<256, 160, 0, stream>>>(xres, nodes, P(base + LP_G1W));

        k_ln<<<256, 160, 0, stream>>>(nodes, P(base + LP_LN2G), P(base + LP_LN2B), xln_bf);
        gemm(xln_bf, 160, 0, lw_bf[l][3], 160, 0, nullptr, 0, 0,
             ff1_bf, 640, 0, P(base + LP_F1B), nullptr, 0, 256, 640, 160, 1, 2,
             nullptr, nullptr, nullptr, 0.f);
        gemm(ff1_bf, 640, 0, lw_bf[l][4], 640, 0, xres, 160, 0, nullptr, 0, 0,
             P(base + LP_F2B), nullptr, 0, 256, 160, 640, 1, 0,
             nullptr, nullptr, nullptr, 0.f);
        k_gate<<<256, 160, 0, stream>>>(xres, nodes, P(base + LP_G2W));
    }

    // ---- head ----
    k_head<<<1, 256, 0, stream>>>(nodes, P(I_FC1W), P(I_FC1B), out_logits);
}
